// CrfDecoder_43026982371872
// MI455X (gfx1250) — matrix operand order repaired and checked by reading, not compiled, not run on hardware
//
#include <hip/hip_runtime.h>

// CRF log-partition forward scan on gfx1250.
// Exp-domain reformulation: the per-step 32x32 log-semiring matmul becomes a
// real bf16 matmul on v_wmma_f32_16x16x32_bf16 (2 WMMAs cover N=0..31).
//
//   true chart[k] = ln2 * ( ls2 + log2 p[k] )      (per-row scale ls2)
//   step:  s = p @ exp(Trans)                       <- WMMA
//          p <- s * exp(e_t)   (rows with t < length only; frozen rows
//                               redirect their store to a trash LDS row)
//   every 4th step: p /= max(p); ls2 += log2(max)   (range control only;
//          exact regardless of cadence since ls2+log2 p is invariant)
//   out[b] = ln2 * ( ls2 + log2( sum_j p[j]*exp(last[j]) ) )
//
// One wave (wave32) owns a 16-row batch tile; grid = 512/16 = 32 waves.
// Scan bounded by the tile's max length (scalar via readfirstlane so EXEC
// stays all-ones for WMMA).

#define BATCH 512
#define TLEN  1024
#define KTAGS 32
#define RS    36    // LDS row stride in floats: 144 B -> 16 B-aligned rows
#define TRASH 16    // extra chart row absorbing stores of frozen rows

typedef __attribute__((ext_vector_type(16))) __bf16 v16bf;
typedef __attribute__((ext_vector_type(8)))  float  v8f;

union ABits { v16bf v; unsigned d[8]; unsigned short u[16]; };

__device__ __forceinline__ unsigned short f2bf_rne(float f) {
    unsigned u = __float_as_uint(f);
    u += 0x7FFFu + ((u >> 16) & 1u);
    return (unsigned short)(u >> 16);
}
// {bf16(hi),bf16(lo)} by truncation: single v_perm_b32
__device__ __forceinline__ unsigned packbf(float lo, float hi) {
    return __builtin_amdgcn_perm(__float_as_uint(hi), __float_as_uint(lo),
                                 0x07060302u);
}
__device__ __forceinline__ float redmax16(float x) {
#pragma unroll
    for (int m = 1; m < 16; m <<= 1) x = fmaxf(x, __shfl_xor(x, m, 32));
    return x;
}
__device__ __forceinline__ float redsum16(float x) {
#pragma unroll
    for (int m = 1; m < 16; m <<= 1) x += __shfl_xor(x, m, 32);
    return x;
}

// One scan step, fully branchless. RESCALE=true additionally renormalizes
// rows (batched cross-lane max so the ds_bpermute waits overlap).
template <bool RESCALE>
__device__ __forceinline__ void crf_step(
    int tq, float* chart, const float* ep0, const float* pfb,
    const ABits& b0, const ABits& b1,
    const int* lenr, float* ls, int grp, int ln)
{
    int tc = tq < TLEN ? tq : TLEN - 1;      // clamp overshoot loads in-bounds
    const float* ept = ep0 + tc * KTAGS;     // r offsets fold into 24-bit imm

    // Issue all emission loads up-front (clause), overlapping gather + WMMA.
    float e0[8], e1[8];
#pragma unroll
    for (int r = 0; r < 8; ++r) {
        e0[r] = ept[r * (TLEN * KTAGS)];
        e1[r] = ept[r * (TLEN * KTAGS) + 16];
    }
    // Prefetch next chunk's emissions: one instruction covers the tile
    // (lane = one 64B half of one row's [t+4] block).
    int tp = (tq + 4) < TLEN ? (tq + 4) : TLEN - 1;
    __builtin_prefetch(pfb + tp * KTAGS, 0, 1);

    // A gather: row = ln, K = 8*grp+[0,8) and 16+8*grp+[0,8)  (4x ds_load_b128)
    const float4* ra = (const float4*)(chart + ln * RS + 8 * grp);
    const float4* rb = (const float4*)(chart + ln * RS + 16 + 8 * grp);
    float4 q0 = ra[0], q1 = ra[1], q2 = rb[0], q3 = rb[1];
    ABits a;
    a.d[0] = packbf(q0.x, q0.y); a.d[1] = packbf(q0.z, q0.w);
    a.d[2] = packbf(q1.x, q1.y); a.d[3] = packbf(q1.z, q1.w);
    a.d[4] = packbf(q2.x, q2.y); a.d[5] = packbf(q2.z, q2.w);
    a.d[6] = packbf(q3.x, q3.y); a.d[7] = packbf(q3.z, q3.w);
    __syncthreads();                          // reads done before writeback

    v8f cz = {};
    v8f c0 = __builtin_amdgcn_wmma_f32_16x16x32_bf16(
                 false, a.v, false, b0.v, (short)0, cz, false, false);
    v8f c1 = __builtin_amdgcn_wmma_f32_16x16x32_bf16(
                 false, a.v, false, b1.v, (short)0, cz, false, false);

    if (!RESCALE) {
#pragma unroll
        for (int r = 0; r < 8; ++r) {
            float v0 = c0[r] * __expf(e0[r]);
            float v1 = c1[r] * __expf(e1[r]);
            // frozen rows store into the trash row: no EXEC masking needed
            int row = (tq < lenr[r]) ? (r + 8 * grp) : TRASH;
            chart[row * RS + ln]      = v0;
            chart[row * RS + ln + 16] = v1;
        }
    } else {
        float v0[8], v1[8], mx[8];
#pragma unroll
        for (int r = 0; r < 8; ++r) {
            v0[r] = c0[r] * __expf(e0[r]);
            v1[r] = c1[r] * __expf(e1[r]);
            mx[r] = fmaxf(v0[r], v1[r]);
        }
#pragma unroll
        for (int lvl = 1; lvl < 16; lvl <<= 1) {   // batched: waits overlap
            float tsh[8];
#pragma unroll
            for (int r = 0; r < 8; ++r) tsh[r] = __shfl_xor(mx[r], lvl, 32);
#pragma unroll
            for (int r = 0; r < 8; ++r) mx[r] = fmaxf(mx[r], tsh[r]);
        }
#pragma unroll
        for (int r = 0; r < 8; ++r) {
            bool alive = tq < lenr[r];
            float inv = __builtin_amdgcn_rcpf(mx[r]);
            int row = alive ? (r + 8 * grp) : TRASH;
            chart[row * RS + ln]      = v0[r] * inv;
            chart[row * RS + ln + 16] = v1[r] * inv;
            ls[r] += alive ? __log2f(mx[r]) : 0.0f;
        }
    }
    __syncthreads();                          // new chart visible to gather
}

__global__ __launch_bounds__(32) void crf_partition_wmma(
    const float* __restrict__ emis,    // [B,T,K]
    const int*   __restrict__ lens,    // [B]
    const float* __restrict__ trans,   // [K,K]
    const float* __restrict__ head,    // [K]
    const float* __restrict__ lastt,   // [K]
    float*       __restrict__ out)     // [B]
{
    __shared__ __align__(16) float chart[17 * RS];   // row 16 = trash row

    const int l   = threadIdx.x;       // 0..31
    const int grp = l >> 4;
    const int ln  = l & 15;
    const int rowBase = blockIdx.x * 16;

    // B operands: exp(Trans) in ISA 32x16 bf16 B-layout (RNE, built once).
    ABits b0, b1;
#pragma unroll
    for (int s = 0; s < 16; ++s) {
        int k = (s < 8) ? (8 * grp + s) : (16 + 8 * grp + (s - 8));
        b0.u[s] = f2bf_rne(__expf(trans[k * KTAGS + ln]));        // N 0..15
        b1.u[s] = f2bf_rne(__expf(trans[k * KTAGS + 16 + ln]));   // N 16..31
    }

    const float* ep0 =
        emis + (size_t)(rowBase + 8 * grp) * (TLEN * KTAGS) + ln;
    // prefetch base: lane covers row (rowBase+ln), 64B half selected by grp
    const float* pfb =
        emis + (size_t)(rowBase + ln) * (TLEN * KTAGS) + grp * 16;

    int lenr[8]; float ls[8];
    int tmax = 1;
#pragma unroll
    for (int r = 0; r < 8; ++r) {
        lenr[r] = lens[rowBase + r + 8 * grp];
        tmax = max(tmax, lenr[r]);
    }
    tmax = max(tmax, __shfl_xor(tmax, 16, 32));   // combine lane halves
    tmax = __builtin_amdgcn_readfirstlane(tmax);  // scalar: EXEC stays full

    // t = 0: chart0 = head + emis[:,0,:], normalized; ls2 in log2 units.
    const float LOG2E = 1.4426950408889634f;
    float h0 = head[ln], h1 = head[ln + 16];
#pragma unroll
    for (int r = 0; r < 8; ++r) {
        float v0 = h0 + ep0[r * (TLEN * KTAGS)];
        float v1 = h1 + ep0[r * (TLEN * KTAGS) + 16];
        float mx = redmax16(fmaxf(v0, v1));
        int m = r + 8 * grp;
        chart[m * RS + ln]      = __expf(v0 - mx);
        chart[m * RS + ln + 16] = __expf(v1 - mx);
        ls[r] = mx * LOG2E;
    }
    __syncthreads();

    // 4x unrolled scan: 3 plain steps + 1 rescale step (range control).
    for (int t = 1; t < tmax; t += 4) {
        crf_step<false>(t,     chart, ep0, pfb, b0, b1, lenr, ls, grp, ln);
        crf_step<false>(t + 1, chart, ep0, pfb, b0, b1, lenr, ls, grp, ln);
        crf_step<false>(t + 2, chart, ep0, pfb, b0, b1, lenr, ls, grp, ln);
        crf_step<true >(t + 3, chart, ep0, pfb, b0, b1, lenr, ls, grp, ln);
    }

    // out = ln2 * (ls2 + log2( sum_j p[j]*exp(last[j]) ))
    const float LN2 = 0.6931471805599453f;
    float e0 = __expf(lastt[ln]), e1 = __expf(lastt[ln + 16]);
#pragma unroll
    for (int r = 0; r < 8; ++r) {
        int m = r + 8 * grp;
        float s = chart[m * RS + ln] * e0 + chart[m * RS + ln + 16] * e1;
        s = redsum16(s);
        if (ln == 0) out[rowBase + m] = LN2 * (ls[r] + __log2f(s));
    }
}

extern "C" void kernel_launch(void* const* d_in, const int* in_sizes, int n_in,
                              void* d_out, int out_size, void* d_ws, size_t ws_size,
                              hipStream_t stream) {
    const float* emis  = (const float*)d_in[0];   // emissions [B,T,K] f32
    const int*   lens  = (const int*)  d_in[1];   // lengths   [B]
    const float* trans = (const float*)d_in[2];   // transitions [K,K]
    const float* head  = (const float*)d_in[3];   // head_transitions [K]
    const float* lastt = (const float*)d_in[4];   // last_transitions [K]
    float* out = (float*)d_out;                   // [B] f32

    crf_partition_wmma<<<BATCH / 16, 32, 0, stream>>>(
        emis, lens, trans, head, lastt, out);
}
